// CompressedAttention_35785667510438
// MI455X (gfx1250) — compile-verified, matrix operand order checked by reading, not run on hardware
//
#include <hip/hip_runtime.h>

// ---------------- problem constants (match reference) ----------------
#define BB    2
#define HH    16
#define KVH_  8
#define DD    128
#define TW    256
#define TCMP  4096
#define TM    8192
#define CC    2048
#define NSEL  1024
#define TOUT  5120
#define GROUPS (HH / KVH_)          // 2
#define SCALE_F 0.08838834764831845f  // 1/sqrt(128)

typedef __attribute__((ext_vector_type(16))) _Float16 v16h;
typedef __attribute__((ext_vector_type(8)))  float    v8f;

// Load one lane's 16-half fragment from f32 memory with the CDNA5 16-bit
// striping: elements e=0..7 -> p[0..7], e=8..15 -> p[16..23].
__device__ __forceinline__ v16h load_frag16(const float* __restrict__ p) {
  const float4* q = (const float4*)p;
  float4 f0 = q[0], f1 = q[1], f2 = q[4], f3 = q[5];
  v16h r;
  r[0]  = (_Float16)f0.x; r[1]  = (_Float16)f0.y; r[2]  = (_Float16)f0.z; r[3]  = (_Float16)f0.w;
  r[4]  = (_Float16)f1.x; r[5]  = (_Float16)f1.y; r[6]  = (_Float16)f1.z; r[7]  = (_Float16)f1.w;
  r[8]  = (_Float16)f2.x; r[9]  = (_Float16)f2.y; r[10] = (_Float16)f2.z; r[11] = (_Float16)f2.w;
  r[12] = (_Float16)f3.x; r[13] = (_Float16)f3.y; r[14] = (_Float16)f3.z; r[15] = (_Float16)f3.w;
  return r;
}

__device__ __forceinline__ void load_bfrag(v16h bf[4], const float* __restrict__ row, int hlf) {
#pragma unroll
  for (int ds = 0; ds < 4; ++ds) bf[ds] = load_frag16(row + ds * 32 + hlf * 8);
}

// S[k,q] 16x16 tile for one lane: A = K rows (16 x 128d), B = Q^T (128d x 16q)
__device__ __forceinline__ v8f score_tile(const float* __restrict__ krow,
                                          const v16h bfrag[4], int hlf) {
  v8f acc = {};
#pragma unroll
  for (int ds = 0; ds < 4; ++ds) {
    v16h a = load_frag16(krow + ds * 32 + hlf * 8);
    acc = __builtin_amdgcn_wmma_f32_16x16x32_f16(false, a, false, bfrag[ds],
                                                 (short)0, acc, false, false);
  }
  return acc;
}

// ---------------- Kernel 1: per-row softmax stats (m, 1/l) -----------------
// grid = B*H*(TW/16) = 512 blocks, 128 threads (4 waves). Wave w strides k-tiles.
__global__ __launch_bounds__(128) void nsa_ml_kernel(
    const float* __restrict__ q_w, const float* __restrict__ km,
    float* __restrict__ mArr, float* __restrict__ lInvArr) {
  const int blk = blockIdx.x;
  const int qt = blk & 15, h = (blk >> 4) & (HH - 1), b = blk >> 8;
  const int kvh = h / GROUPS;
  const int tid = threadIdx.x;
  const int lane = tid & 31, wave = tid >> 5;
  const int lq = lane & 15, hlf = lane >> 4;

  v16h bfrag[4];
  load_bfrag(bfrag, q_w + (((size_t)(b * HH + h)) * TW + (qt * 16 + lq)) * DD, hlf);

  const float* kbase = km + ((size_t)(b * KVH_ + kvh)) * TCMP * DD;
  float mrun = -__builtin_inff(), lrun = 0.0f;
  for (int kt = wave * 16; kt < TCMP; kt += 64) {
    v8f acc = score_tile(kbase + (size_t)(kt + lq) * DD, bfrag, hlf);
    float tmax = mrun;
#pragma unroll
    for (int r = 0; r < 8; ++r) tmax = fmaxf(tmax, acc[r] * SCALE_F);
    lrun *= __expf(mrun - tmax);
#pragma unroll
    for (int r = 0; r < 8; ++r) lrun += __expf(acc[r] * SCALE_F - tmax);
    mrun = tmax;
  }
  // merge the two 16-lane halves (same q column, disjoint k rows)
  float mo = __shfl_xor(mrun, 16, 32);
  float lo = __shfl_xor(lrun, 16, 32);
  float mm = fmaxf(mrun, mo);
  float ll = lrun * __expf(mrun - mm) + lo * __expf(mo - mm);

  __shared__ float mP[4][16], lP[4][16];
  if (lane < 16) { mP[wave][lane] = mm; lP[wave][lane] = ll; }
  __syncthreads();
  if (tid < 16) {
    float M = mP[0][tid], L = lP[0][tid];
#pragma unroll
    for (int w = 1; w < 4; ++w) {
      float m2 = mP[w][tid], l2 = lP[w][tid];
      float mn = fmaxf(M, m2);
      L = L * __expf(M - mn) + l2 * __expf(m2 - mn);
      M = mn;
    }
    size_t o = ((size_t)(b * HH + h)) * TW + qt * 16 + tid;
    mArr[o] = M;
    lInvArr[o] = 1.0f / L;
  }
}

// ------------- Kernel 2: importance partials (no atomics, k-owned) ----------
// grid = B * (TCMP/64) * KVH = 1024 blocks, 128 threads. Wave owns 16 k rows.
// part[((b*KVH+kvh)*TCMP)+k] = sum over (2 heads x 256 q) of attn prob.
__global__ __launch_bounds__(128) void nsa_importance_kernel(
    const float* __restrict__ q_w, const float* __restrict__ km,
    const float* __restrict__ mArr, const float* __restrict__ lInvArr,
    float* __restrict__ part) {
  const int blk = blockIdx.x;
  const int kvh = blk & 7, chunk = (blk >> 3) & 63, b = blk >> 9;
  const int tid = threadIdx.x;
  const int lane = tid & 31, wave = tid >> 5;
  const int lq = lane & 15, hlf = lane >> 4;
  const int kt = chunk * 64 + wave * 16;

  float colsum[8];
#pragma unroll
  for (int r = 0; r < 8; ++r) colsum[r] = 0.0f;

  // K-side fragments are loop-invariant for this block
  const float* krow = km + (((size_t)(b * KVH_ + kvh)) * TCMP + kt + lq) * DD;
  v16h afrag[4];
#pragma unroll
  for (int ds = 0; ds < 4; ++ds) afrag[ds] = load_frag16(krow + ds * 32 + hlf * 8);

  for (int g = 0; g < GROUPS; ++g) {
    const int h = kvh * GROUPS + g;
    const float* qb  = q_w + ((size_t)(b * HH + h)) * TW * DD;
    const float* mb  = mArr + ((size_t)(b * HH + h)) * TW;
    const float* lib = lInvArr + ((size_t)(b * HH + h)) * TW;
    for (int qt = 0; qt < 16; ++qt) {
      v16h bfrag[4];
      load_bfrag(bfrag, qb + (size_t)(qt * 16 + lq) * DD, hlf);
      float mq = mb[qt * 16 + lq];
      float li = lib[qt * 16 + lq];
      v8f acc = {};
#pragma unroll
      for (int ds = 0; ds < 4; ++ds)
        acc = __builtin_amdgcn_wmma_f32_16x16x32_f16(false, afrag[ds], false, bfrag[ds],
                                                     (short)0, acc, false, false);
#pragma unroll
      for (int r = 0; r < 8; ++r) {
        float p = __expf(acc[r] * SCALE_F - mq) * li;   // softmax prob, this lane's q
        p += __shfl_xor(p, 1, 16);                      // sum over the 16 q columns
        p += __shfl_xor(p, 2, 16);
        p += __shfl_xor(p, 4, 16);
        p += __shfl_xor(p, 8, 16);
        colsum[r] += p;
      }
    }
  }
  if (lq == 0) {                                        // lanes 0 and 16 own 8 k's each
#pragma unroll
    for (int r = 0; r < 8; ++r)
      part[((size_t)(b * KVH_ + kvh)) * TCMP + kt + r + hlf * 8] = colsum[r];
  }
}

// ---------- Kernel 3: per-batch top-k + exclusive scan of sizes -------------
__global__ __launch_bounds__(1024) void nsa_topk_scan_kernel(
    const float* __restrict__ part,
    int* __restrict__ startArr, int* __restrict__ maskArr) {
  const int b = blockIdx.x;
  const int t = threadIdx.x;
  __shared__ float vals[TCMP];
  __shared__ int   idxs[TCMP];
  __shared__ int   msk[TCMP];
  __shared__ int   psum[1024];

  for (int i = t; i < TCMP; i += 1024) {
    float s = 0.0f;
#pragma unroll
    for (int kv = 0; kv < KVH_; ++kv)
      s += part[((size_t)(b * KVH_ + kv)) * TCMP + i];
    vals[i] = s * (1.0f / HH);
    idxs[i] = i;
  }
  __syncthreads();

  // bitonic sort: descending value, tie -> lower index first (matches top_k)
  for (int k2 = 2; k2 <= TCMP; k2 <<= 1) {
    for (int j = k2 >> 1; j > 0; j >>= 1) {
      for (int i = t; i < TCMP; i += 1024) {
        int ixj = i ^ j;
        if (ixj > i) {
          float v1 = vals[i], v2 = vals[ixj];
          int   i1 = idxs[i], i2 = idxs[ixj];
          bool oo  = (v2 > v1) || (v2 == v1 && i2 < i1);  // ixj should precede i
          bool dir = ((i & k2) == 0);
          if (oo == dir) { vals[i] = v2; vals[ixj] = v1; idxs[i] = i2; idxs[ixj] = i1; }
        }
      }
      __syncthreads();
    }
  }

  for (int i = t; i < TCMP; i += 1024) msk[i] = 0;
  __syncthreads();
  msk[idxs[t]] = 1;                 // ranks 0..NSEL-1 (NSEL == blockDim == 1024)
  __syncthreads();

  // exclusive scan of sizes = 1 + selected, 4 elements per thread
  const int base = t * 4;
  int s = 0;
  int loc0 = s; s += 1 + msk[base + 0];
  int loc1 = s; s += 1 + msk[base + 1];
  int loc2 = s; s += 1 + msk[base + 2];
  int loc3 = s; s += 1 + msk[base + 3];
  psum[t] = s;
  __syncthreads();
  for (int off = 1; off < 1024; off <<= 1) {
    int v = (t >= off) ? psum[t - off] : 0;
    __syncthreads();
    psum[t] += v;
    __syncthreads();
  }
  int pre = (t == 0) ? 0 : psum[t - 1];
  size_t o = (size_t)b * TCMP + base;
  startArr[o + 0] = pre + loc0;  startArr[o + 1] = pre + loc1;
  startArr[o + 2] = pre + loc2;  startArr[o + 3] = pre + loc3;
  maskArr[o + 0] = msk[base + 0]; maskArr[o + 1] = msk[base + 1];
  maskArr[o + 2] = msk[base + 2]; maskArr[o + 3] = msk[base + 3];
}

// -------- Kernel 4: scatter/expand (every output row written once) ----------
__global__ __launch_bounds__(256) void nsa_scatter_kernel(
    const float* __restrict__ x_m, const float* __restrict__ xm_cmp,
    const int* __restrict__ startArr, const int* __restrict__ maskArr,
    float* __restrict__ y) {
  const int blk = blockIdx.x;                 // b*TCMP + k
  const int b = blk >> 12, k = blk & (TCMP - 1);
  const int st = startArr[blk];
  const int sel = maskArr[blk];
  const int t = threadIdx.x;
  float4* dst = (float4*)(y + ((size_t)b * TOUT + st) * CC);
  if (!sel) {
    const float4* src = (const float4*)(xm_cmp + ((size_t)b * TCMP + k) * CC);
    for (int i = t; i < CC / 4; i += 256) dst[i] = src[i];
  } else {
    // rows 2k and 2k+1 are contiguous in x_m; rows st, st+1 contiguous in y
    const float4* src = (const float4*)(x_m + ((size_t)b * TM + 2 * k) * CC);
    for (int i = t; i < (2 * CC) / 4; i += 256) dst[i] = src[i];
  }
}

// ---------------------------------------------------------------------------
extern "C" void kernel_launch(void* const* d_in, const int* in_sizes, int n_in,
                              void* d_out, int out_size, void* d_ws, size_t ws_size,
                              hipStream_t stream) {
  const float* q_w    = (const float*)d_in[0];   // [B,H,TW,D]
  const float* km     = (const float*)d_in[1];   // [B,KVH,TCMP,D]
  const float* x_m    = (const float*)d_in[2];   // [B,TM,C]
  const float* xm_cmp = (const float*)d_in[3];   // [B,TCMP,C]
  float* y = (float*)d_out;                      // [B,TOUT,C]

  float* mArr  = (float*)d_ws;                   // B*H*TW      = 8192
  float* lInv  = mArr + (size_t)BB * HH * TW;    // 8192
  float* part  = lInv + (size_t)BB * HH * TW;    // B*KVH*TCMP  = 65536
  int* startA  = (int*)(part + (size_t)BB * KVH_ * TCMP);  // 8192
  int* maskA   = startA + (size_t)BB * TCMP;               // 8192
  // total workspace: 384 KB

  nsa_ml_kernel<<<BB * HH * (TW / 16), 128, 0, stream>>>(q_w, km, mArr, lInv);
  nsa_importance_kernel<<<BB * (TCMP / 64) * KVH_, 128, 0, stream>>>(q_w, km, mArr, lInv, part);
  nsa_topk_scan_kernel<<<BB, 1024, 0, stream>>>(part, startA, maskA);
  nsa_scatter_kernel<<<BB * TCMP, 256, 0, stream>>>(x_m, xm_cmp, startA, maskA, y);
}
